// CoSTCoModel_22136261443933
// MI455X (gfx1250) — matrix operand order repaired and checked by reading, not compiled, not run on hardware
//
#include <hip/hip_runtime.h>
#include <hip/hip_bf16.h>

// ---------------------------------------------------------------------------
// CoSTCo fused forward for MI455X (gfx1250, wave32, WMMA 16x16x32 f16).
//
// B=131072, RANK=128, C=256, MODES=3.  ~95 GFLOP; weights L2-resident.
// Bottleneck is L2->WGP weight streaming: each 16-row M-tile re-reads the
// weights, so each wave applies every fetched B fragment to NRT=4 row-tiles
// (weight traffic /4 vs one-row-tile-per-wave).  A fragments come from LDS.
// BM=64 rows/block, 512 threads = 16 waves, wave w owns columns w*16..+15.
// ---------------------------------------------------------------------------

typedef __attribute__((ext_vector_type(16))) _Float16 v16h;
typedef __attribute__((ext_vector_type(8)))  _Float16 v8h;
typedef __attribute__((ext_vector_type(4)))  _Float16 v4h;
typedef __attribute__((ext_vector_type(8)))  float    v8f;

union V16 { v16h v; v8h h[2]; };

#define RANK 128
#define NCH  256
#define NM   3
#define BM   64            // batch rows per workgroup (4 WMMA M-tiles)
#define NRT  4             // row tiles per wave (B-fragment reuse factor)
#define NTHREADS 512       // 16 waves; wave w -> columns w*16..w*16+15
#define H1STR 776          // 768 + 8 halfs pad (1552B, 16B aligned)
#define H2STR 264          // 256 + 8 halfs pad (528B, 16B aligned)

// dynamic-LDS partition (in halfs)
#define XS_OFF 0
#define XS_SZ  (NM * BM * RANK)          // 24576
#define H1_OFF (XS_OFF + XS_SZ)
#define H1_SZ  (BM * H1STR)              // 49664
#define H2_OFF (H1_OFF + H1_SZ)
#define H2_SZ  (BM * H2STR)              // 16896
#define H3_OFF (H2_OFF + H2_SZ)
#define H3_SZ  (BM * H2STR)              // 16896
#define LDS_HALFS (H3_OFF + H3_SZ)       // 108032 halfs = 216064 B (<320KB)

#define N_W1   (NCH*RANK)        // 32768
#define N_W2   (NCH*NCH*NM)      // 196608
#define N_WFC1 (NCH*NCH)         // 65536
#define N_WFC2 (NCH)             // 256

// --- fp32 -> f16 weight conversion (runs inside the graph, deterministic) ---
__global__ void costco_cvt_weights(const float* __restrict__ w1,
                                   const float* __restrict__ w2,
                                   const float* __restrict__ wfc1,
                                   const float* __restrict__ wfc2,
                                   _Float16* __restrict__ w1h,
                                   _Float16* __restrict__ w2h,
                                   _Float16* __restrict__ wfc1h,
                                   _Float16* __restrict__ wfc2h) {
    int i = blockIdx.x * blockDim.x + threadIdx.x;
    if (i < N_W1) w1h[i] = (_Float16)w1[i];
    int j = i - N_W1;
    if (j >= 0 && j < N_W2) w2h[j] = (_Float16)w2[j];
    int k = j - N_W2;
    if (k >= 0 && k < N_WFC1) wfc1h[k] = (_Float16)wfc1[k];
    int l = k - N_WFC1;
    if (l >= 0 && l < N_WFC2) wfc2h[l] = (_Float16)wfc2[l];
}

// --- fused forward: 64-row batch tile per 512-thread (16-wave) block --------
__global__ __launch_bounds__(NTHREADS, 1)
void costco_fused(const int*    __restrict__ idx,     // (B,3)
                  const float*  __restrict__ e0,      // (339,128)
                  const float*  __restrict__ e1,      // (5825,128)
                  const float*  __restrict__ e2,      // (64,128)
                  const _Float16* __restrict__ w1h,   // (256,128)
                  const float*  __restrict__ b1,
                  const _Float16* __restrict__ w2h,   // (256, 768) k=c*3+m
                  const float*  __restrict__ b2,
                  const _Float16* __restrict__ wfc1h, // (256,256)
                  const float*  __restrict__ bfc1,
                  const _Float16* __restrict__ wfc2h, // (256)
                  const float*  __restrict__ bfc2,
                  float* __restrict__ out) {
    extern __shared__ __align__(16) char smem_raw[];
    _Float16* smem = (_Float16*)smem_raw;
    _Float16* Xs  = smem + XS_OFF;
    _Float16* H1s = smem + H1_OFF;
    _Float16* H2s = smem + H2_OFF;
    _Float16* H3s = smem + H3_OFF;
    __shared__ int sIdx[BM * NM];                     // 768 B static

    const int t  = threadIdx.x;
    const int b0 = blockIdx.x * BM;

    if (t < BM * NM) {
        int r = t / NM, m = t % NM;
        sIdx[t] = idx[(size_t)(b0 + r) * NM + m];
    }
    __syncthreads();

    // ---- gather embeddings -> LDS (f16), cooperative float4 loads ----
    for (int i = t; i < NM * BM * (RANK / 4); i += NTHREADS) {  // 6144 chunks
        int rm = i >> 5;            // (m*64 + r), 0..191
        int ch = i & 31;            // float4 chunk within the 128-wide row
        int m  = rm / BM;
        int r  = rm % BM;
        const float* ep = (m == 0) ? e0 : ((m == 1) ? e1 : e2);
        const float4 v = *(const float4*)(ep + (size_t)sIdx[r * NM + m] * RANK + ch * 4);
        v4h h = { (_Float16)v.x, (_Float16)v.y, (_Float16)v.z, (_Float16)v.w };
        *(v4h*)&Xs[rm * RANK + ch * 4] = h;
    }
    __syncthreads();

    const int lane   = t & 31;
    const int wv     = t >> 5;       // 16 waves
    const int lanelo = lane & 15;    // A row or B/C column within tile
    const int hi     = lane >> 4;    // K-half selector per ISA fragment layout
    const int col    = wv * 16 + lanelo;   // this wave's fixed output column

    // prefetch this wave's stage-2 weight row while stage 1 computes
    __builtin_prefetch(w2h + (size_t)col * (NCH * NM), 0, 1);

    // ================= stage 1: h1[b,c,m] = relu(X_m @ W1^T + b1) ==========
    {
        const float bias = b1[col];
        for (int m = 0; m < NM; m++) {
            v8f acc[NRT] = {};
            #pragma unroll
            for (int ks = 0; ks < RANK / 32; ks++) {       // 4 k-steps
                V16 b;
                const _Float16* wp = w1h + (size_t)col * RANK + ks * 32 + hi * 8;
                b.h[0] = *(const v8h*)wp;
                b.h[1] = *(const v8h*)(wp + 16);
                #pragma unroll
                for (int rt = 0; rt < NRT; rt++) {
                    V16 a;
                    const int ab =
                        (m * BM + rt * 16 + lanelo) * RANK + ks * 32 + hi * 8;
                    a.h[0] = *(const v8h*)&Xs[ab];
                    a.h[1] = *(const v8h*)&Xs[ab + 16];
                    acc[rt] = __builtin_amdgcn_wmma_f32_16x16x32_f16(
                                  false, a.v, false, b.v, (short)0, acc[rt],
                                  false, false);
                }
            }
            #pragma unroll
            for (int rt = 0; rt < NRT; rt++) {
                #pragma unroll
                for (int i = 0; i < 8; i++) {
                    float v = acc[rt][i] + bias;
                    v = v > 0.f ? v : 0.f;
                    const int row = rt * 16 + i + hi * 8;
                    H1s[row * H1STR + col * NM + m] = (_Float16)v;  // k=c*3+m
                }
            }
        }
    }
    __syncthreads();

    // ================= stage 2: h2[b,d] = relu(H1 @ W2^T + b2), K=768 ======
    {
        const float bias = b2[col];
        v8f acc[NRT] = {};
        #pragma unroll
        for (int ks = 0; ks < (NCH * NM) / 32; ks++) {     // 24 k-steps
            V16 b;
            const _Float16* wp =
                w2h + (size_t)col * (NCH * NM) + ks * 32 + hi * 8;
            b.h[0] = *(const v8h*)wp;
            b.h[1] = *(const v8h*)(wp + 16);
            #pragma unroll
            for (int rt = 0; rt < NRT; rt++) {
                V16 a;
                const int ab = (rt * 16 + lanelo) * H1STR + ks * 32 + hi * 8;
                a.h[0] = *(const v8h*)&H1s[ab];
                a.h[1] = *(const v8h*)&H1s[ab + 16];
                acc[rt] = __builtin_amdgcn_wmma_f32_16x16x32_f16(
                              false, a.v, false, b.v, (short)0, acc[rt],
                              false, false);
            }
        }
        #pragma unroll
        for (int rt = 0; rt < NRT; rt++) {
            #pragma unroll
            for (int i = 0; i < 8; i++) {
                float v = acc[rt][i] + bias;
                v = v > 0.f ? v : 0.f;
                H2s[(rt * 16 + i + hi * 8) * H2STR + col] = (_Float16)v;
            }
        }
    }
    __syncthreads();

    // ================= stage 3: h3[b,e] = relu(H2 @ Wfc1^T + bfc1), K=256 ==
    {
        const float bias = bfc1[col];
        v8f acc[NRT] = {};
        #pragma unroll
        for (int ks = 0; ks < NCH / 32; ks++) {            // 8 k-steps
            V16 b;
            const _Float16* wp = wfc1h + (size_t)col * NCH + ks * 32 + hi * 8;
            b.h[0] = *(const v8h*)wp;
            b.h[1] = *(const v8h*)(wp + 16);
            #pragma unroll
            for (int rt = 0; rt < NRT; rt++) {
                V16 a;
                const int ab = (rt * 16 + lanelo) * H2STR + ks * 32 + hi * 8;
                a.h[0] = *(const v8h*)&H2s[ab];
                a.h[1] = *(const v8h*)&H2s[ab + 16];
                acc[rt] = __builtin_amdgcn_wmma_f32_16x16x32_f16(
                              false, a.v, false, b.v, (short)0, acc[rt],
                              false, false);
            }
        }
        #pragma unroll
        for (int rt = 0; rt < NRT; rt++) {
            #pragma unroll
            for (int i = 0; i < 8; i++) {
                float v = acc[rt][i] + bias;
                v = v > 0.f ? v : 0.f;
                H3s[(rt * 16 + i + hi * 8) * H2STR + col] = (_Float16)v;
            }
        }
    }
    __syncthreads();

    // ================= stage 4: out[b] = relu(H3 . wfc2 + bfc2) ===========
    if (t < BM) {
        float acc = bfc2[0];
        #pragma unroll 8
        for (int e = 0; e < NCH; e++)
            acc += (float)H3s[t * H2STR + e] * (float)wfc2h[e];
        out[b0 + t] = acc > 0.f ? acc : 0.f;
    }
}

extern "C" void kernel_launch(void* const* d_in, const int* in_sizes, int n_in,
                              void* d_out, int out_size, void* d_ws, size_t ws_size,
                              hipStream_t stream) {
    const int*   idx  = (const int*)  d_in[0];
    const float* e0   = (const float*)d_in[1];
    const float* e1   = (const float*)d_in[2];
    const float* e2   = (const float*)d_in[3];
    const float* w1   = (const float*)d_in[4];
    const float* b1   = (const float*)d_in[5];
    const float* w2   = (const float*)d_in[6];
    const float* b2   = (const float*)d_in[7];
    const float* wfc1 = (const float*)d_in[8];
    const float* bfc1 = (const float*)d_in[9];
    const float* wfc2 = (const float*)d_in[10];
    const float* bfc2 = (const float*)d_in[11];
    float* out = (float*)d_out;

    _Float16* ws    = (_Float16*)d_ws;
    _Float16* w1h   = ws;
    _Float16* w2h   = w1h + N_W1;
    _Float16* wfc1h = w2h + N_W2;
    _Float16* wfc2h = wfc1h + N_WFC1;

    const int ncvt = N_W1 + N_W2 + N_WFC1 + N_WFC2;
    costco_cvt_weights<<<(ncvt + 255) / 256, 256, 0, stream>>>(
        w1, w2, wfc1, wfc2, w1h, w2h, wfc1h, wfc2h);

    const int nblocks = out_size / BM;               // 131072/64 = 2048
    const size_t lds_bytes = LDS_HALFS * sizeof(_Float16);  // 216064 B
    costco_fused<<<nblocks, NTHREADS, lds_bytes, stream>>>(
        idx, e0, e1, e2, w1h, b1, w2h, b2, wfc1h, bfc1, wfc2h, bfc2, out);
}
